// Attention_7911329759504
// MI455X (gfx1250) — matrix operand order repaired and checked by reading, not compiled, not run on hardware
//
#include <hip/hip_runtime.h>
#include <hip/hip_bf16.h>

typedef unsigned short u16;
typedef __attribute__((ext_vector_type(16))) __bf16 bf16x16;
typedef __attribute__((ext_vector_type(8)))  __bf16 bf16x8;
typedef __attribute__((ext_vector_type(8)))  float  f32x8;

#define NHQ   32
#define NKVH  8
#define HDIM  64
#define SLEN  2048
#define HID   2048
#define QKVN  ((NHQ + 2*NKVH)*HDIM)   // 3072

// round-to-nearest-even f32 -> bf16 (as raw u16)
static __device__ __forceinline__ u16 f2bf(float f) {
  unsigned int u = __float_as_uint(f);
  u = u + 0x7FFFu + ((u >> 16) & 1u);
  return (u16)(u >> 16);
}

static __device__ __forceinline__ bf16x16 cat8(bf16x8 lo, bf16x8 hi) {
  bf16x16 r;
#pragma unroll
  for (int e = 0; e < 8; ++e) { r[e] = lo[e]; r[e + 8] = hi[e]; }
  return r;
}

static __device__ __forceinline__ f32x8 wmma_bf16(bf16x16 a, bf16x16 b, f32x8 c) {
  // (neg_a, A, neg_b, B, c_mod, C, reuse_a, reuse_b)
  return __builtin_amdgcn_wmma_f32_16x16x32_bf16(false, a, false, b, (short)0, c,
                                                 false, false);
}

// ---------------- f32 -> bf16 conversion ----------------
__global__ void k_f32_to_bf16(const float* __restrict__ in, u16* __restrict__ out, int n) {
  int i = blockIdx.x * blockDim.x + threadIdx.x;
  int stride = gridDim.x * blockDim.x;
  for (; i < n; i += stride) out[i] = f2bf(in[i]);
}

// ---------------- C[M,N](f32) = A[M,K](bf16) @ W[N,K]^T(bf16) ----------------
// one wave computes a 32x64 output tile: 2 A-fragments (rows m0..15, m16..31)
// share each of 4 B-fragments -> 8 WMMAs per 32-deep K step.
__global__ __launch_bounds__(32) void k_gemm_bf16_nt(
    const u16* __restrict__ A, const u16* __restrict__ W,
    float* __restrict__ C, int M, int N, int K)
{
  const int lane = threadIdx.x;
  const int half = lane >> 4, l16 = lane & 15;
  const int m0 = blockIdx.x * 32;
  const int n0 = blockIdx.y * 64;

  f32x8 acc[2][4] = {};
  const u16* Arow0 = A + (size_t)(m0 + l16) * K;
  const u16* Arow1 = A + (size_t)(m0 + 16 + l16) * K;
  const u16* Wbase = W + (size_t)n0 * K;

  for (int k0 = 0; k0 < K; k0 += 32) {
    if (k0 + 128 < K) {  // prefetch one iteration group ahead (global_prefetch_b8)
      __builtin_prefetch(Arow0 + k0 + 128, 0, 0);
      __builtin_prefetch(Arow1 + k0 + 128, 0, 0);
      __builtin_prefetch(Wbase + (size_t)l16 * K + k0 + 128, 0, 0);
    }
    // A fragments: lane l16 = row M, two 8-wide contiguous K chunks
    bf16x16 a0 = cat8(*(const bf16x8*)(Arow0 + k0 + 8*half),
                      *(const bf16x8*)(Arow0 + k0 + 16 + 8*half));
    bf16x16 a1 = cat8(*(const bf16x8*)(Arow1 + k0 + 8*half),
                      *(const bf16x8*)(Arow1 + k0 + 16 + 8*half));
#pragma unroll
    for (int nt = 0; nt < 4; ++nt) {
      // B fragment: B[k][n] = W[n][k]; lane reads two 8-wide k chunks (8*half, 16+8*half) from row n
      const u16* Wrow = Wbase + (size_t)(nt*16 + l16) * K;
      bf16x16 b = cat8(*(const bf16x8*)(Wrow + k0 + 8*half), *(const bf16x8*)(Wrow + k0 + 16 + 8*half));
      acc[0][nt] = wmma_bf16(a0, b, acc[0][nt]);
      acc[1][nt] = wmma_bf16(a1, b, acc[1][nt]);
    }
  }
  // C/D layout: element i of acc, lane L -> row = 8*(L/16)+i, col = L%16
#pragma unroll
  for (int mt = 0; mt < 2; ++mt)
#pragma unroll
    for (int nt = 0; nt < 4; ++nt)
#pragma unroll
      for (int i = 0; i < 8; ++i) {
        int m = m0 + mt*16 + half*8 + i;
        int n = n0 + nt*16 + l16;
        C[(size_t)m * N + n] = acc[mt][nt][i];
      }
}

// ---------------- RoPE on q,k + split/convert q,k,v to bf16 ----------------
__global__ __launch_bounds__(32) void k_rope_split(
    const float* __restrict__ qkv, const float* __restrict__ fc,
    u16* __restrict__ qh, u16* __restrict__ kh, u16* __restrict__ vh)
{
  const int s = blockIdx.x, h = blockIdx.y, p = threadIdx.x; // p = rotary pair idx
  const float* row = qkv + (size_t)s * QKVN + (size_t)h * HDIM;
  float x0 = row[2*p], x1 = row[2*p + 1];
  if (h < NHQ) {
    float c = fc[(size_t)s*HDIM + 2*p], sn = fc[(size_t)s*HDIM + 2*p + 1];
    u16* o = qh + (size_t)s * (NHQ*HDIM) + h*HDIM;
    o[2*p]     = f2bf(x0*c - x1*sn);
    o[2*p + 1] = f2bf(x1*c + x0*sn);
  } else if (h < NHQ + NKVH) {
    int hk = h - NHQ;
    float c = fc[(size_t)s*HDIM + 2*p], sn = fc[(size_t)s*HDIM + 2*p + 1];
    u16* o = kh + (size_t)s * (NKVH*HDIM) + hk*HDIM;
    o[2*p]     = f2bf(x0*c - x1*sn);
    o[2*p + 1] = f2bf(x1*c + x0*sn);
  } else {
    int hv = h - NHQ - NKVH;
    u16* o = vh + (size_t)s * (NKVH*HDIM) + hv*HDIM;
    o[2*p]     = f2bf(x0);
    o[2*p + 1] = f2bf(x1);
  }
}

// ---------------- flash attention: one wave per (q head, 16-row query block) ----
__global__ __launch_bounds__(32) void k_flash_attn(
    const u16* __restrict__ qh, const u16* __restrict__ kh,
    const u16* __restrict__ vh, u16* __restrict__ yh)
{
  __shared__ __align__(32) u16 lK[32][64];   // K tile, row = t_local (128B rows)
  __shared__ __align__(32) u16 lVt[64][32];  // V tile transposed, row = d
  __shared__ __align__(32) u16 lP[16][32];   // softmax tile for re-layout

  const int h  = blockIdx.x;       // query head
  const int qb = blockIdx.y * 16;  // query row base
  const int kv = h >> 2;           // GQA: 4 q heads per kv head
  const int lane = threadIdx.x;
  const int half = lane >> 4, l16 = lane & 15;
  const float NEG_INF = -__builtin_inff();

  // Q fragments for the two 32-wide halves of HD=64
  const u16* Qrow = qh + (size_t)(qb + l16) * (NHQ*HDIM) + h*HDIM;
  bf16x16 aQ[2];
#pragma unroll
  for (int kk = 0; kk < 2; ++kk) {
    bf16x8 lo = *(const bf16x8*)(Qrow + kk*32 + 16*half);
    bf16x8 hi = *(const bf16x8*)(Qrow + kk*32 + 16*half + 8);
    aQ[kk] = cat8(lo, hi);
  }

  f32x8 acc[4] = {};
  float mrow[8], lrow[8];
#pragma unroll
  for (int i = 0; i < 8; ++i) { mrow[i] = NEG_INF; lrow[i] = 0.0f; }
  const float sc = 0.125f;  // HD^-0.5

  // per-lane LDS dest (row = lane) and per-lane global byte offset for async copy
  const unsigned ldsK = (unsigned)(unsigned long long)&lK[lane][0];
  const unsigned kvoff = (unsigned)lane * (NKVH*HDIM*2);  // global row stride bytes

  for (int t0 = 0; t0 < qb + 16; t0 += 32) {
    __syncthreads();
    // --- stage K tile via CDNA5 async global->LDS (ASYNCcnt-tracked) ---
    {
      const u16* kbase = kh + (size_t)t0 * (NKVH*HDIM) + kv*HDIM;  // uniform
      asm volatile(
          "global_load_async_to_lds_b128 %0, %1, %2\n\t"
          "global_load_async_to_lds_b128 %0, %1, %2 offset:16\n\t"
          "global_load_async_to_lds_b128 %0, %1, %2 offset:32\n\t"
          "global_load_async_to_lds_b128 %0, %1, %2 offset:48\n\t"
          "global_load_async_to_lds_b128 %0, %1, %2 offset:64\n\t"
          "global_load_async_to_lds_b128 %0, %1, %2 offset:80\n\t"
          "global_load_async_to_lds_b128 %0, %1, %2 offset:96\n\t"
          "global_load_async_to_lds_b128 %0, %1, %2 offset:112"
          :
          : "v"(ldsK), "v"(kvoff), "s"(kbase)
          : "memory");
    }
    // --- stage V transposed (needs lane shuffle; normal loads + ds stores) ---
    {
      const u16* vr = vh + (size_t)(t0 + lane) * (NKVH*HDIM) + kv*HDIM;
#pragma unroll
      for (int d = 0; d < 64; ++d) lVt[d][lane] = vr[d];
    }
    asm volatile("s_wait_asynccnt 0" ::: "memory");
    __syncthreads();

    // scores: S = Q @ K^T as two 16x16 tiles (t 0..15, 16..31), K-dim = 64
    f32x8 s0 = {}, s1 = {};
#pragma unroll
    for (int kk = 0; kk < 2; ++kk) {
      bf16x16 b0 = *(const bf16x16*)&lK[l16][kk*32 + 16*half];
      s0 = wmma_bf16(aQ[kk], b0, s0);
      bf16x16 b1 = *(const bf16x16*)&lK[16 + l16][kk*32 + 16*half];
      s1 = wmma_bf16(aQ[kk], b1, s1);
    }

    // online softmax: rows 8*half+i live in one 16-lane half; reduce via shfl_xor
    float p0[8], p1[8];
#pragma unroll
    for (int i = 0; i < 8; ++i) {
      int srow = qb + half*8 + i;
      int c0 = t0 + l16;
      int c1 = t0 + 16 + l16;
      float v0 = (c0 <= srow) ? s0[i]*sc : NEG_INF;  // causal mask
      float v1 = (c1 <= srow) ? s1[i]*sc : NEG_INF;
      float vm = fmaxf(v0, v1);
#pragma unroll
      for (int off = 1; off < 16; off <<= 1)
        vm = fmaxf(vm, __shfl_xor(vm, off, 32));
      float mnew = fmaxf(mrow[i], vm);
      float scl  = __expf(mrow[i] - mnew);
      mrow[i] = mnew;
      p0[i] = __expf(v0 - mnew);
      p1[i] = __expf(v1 - mnew);
      float rs = p0[i] + p1[i];
#pragma unroll
      for (int off = 1; off < 16; off <<= 1)
        rs += __shfl_xor(rs, off, 32);
      lrow[i] = lrow[i]*scl + rs;
#pragma unroll
      for (int nt = 0; nt < 4; ++nt) acc[nt][i] = acc[nt][i] * scl;
      // stash P (C-layout) into LDS to reload in A-fragment layout
      lP[half*8 + i][l16]      = f2bf(p0[i]);
      lP[half*8 + i][16 + l16] = f2bf(p1[i]);
    }
    __syncthreads();

    // O += P (16x32) @ V (32x64): A from lP, B from transposed V tile
    bf16x8 plo = *(const bf16x8*)&lP[l16][16*half];
    bf16x8 phi = *(const bf16x8*)&lP[l16][16*half + 8];
    bf16x16 aP = cat8(plo, phi);
#pragma unroll
    for (int nt = 0; nt < 4; ++nt) {
      bf16x16 bV = *(const bf16x16*)&lVt[nt*16 + l16][16*half];
      acc[nt] = wmma_bf16(aP, bV, acc[nt]);
    }
  }

  // normalize and write bf16 head output into y[S, HID]
#pragma unroll
  for (int nt = 0; nt < 4; ++nt)
#pragma unroll
    for (int i = 0; i < 8; ++i) {
      int srow = qb + half*8 + i;
      int d = nt*16 + l16;
      yh[(size_t)srow * HID + h*HDIM + d] = f2bf(acc[nt][i] / lrow[i]);
    }
}

extern "C" void kernel_launch(void* const* d_in, const int* in_sizes, int n_in,
                              void* d_out, int out_size, void* d_ws, size_t ws_size,
                              hipStream_t stream) {
  (void)in_sizes; (void)n_in; (void)out_size; (void)ws_size;
  const float* x    = (const float*)d_in[0];
  const float* fc   = (const float*)d_in[1];
  // d_in[2] = causal mask (bool): computed analytically (t <= s), not read
  const float* Wqkv = (const float*)d_in[3];
  const float* Wo   = (const float*)d_in[4];
  float* out = (float*)d_out;

  char* ws = (char*)d_ws;
  size_t off = 0;
  auto wsalloc = [&](size_t bytes) -> void* {
    void* p = ws + off;
    off += (bytes + 255) & ~(size_t)255;
    return p;
  };
  u16*   xh    = (u16*)  wsalloc((size_t)SLEN*HID*2);
  u16*   wqkvh = (u16*)  wsalloc((size_t)QKVN*HID*2);
  u16*   woh   = (u16*)  wsalloc((size_t)HID*HID*2);
  float* qkvf  = (float*)wsalloc((size_t)SLEN*QKVN*4);
  u16*   qh    = (u16*)  wsalloc((size_t)SLEN*NHQ*HDIM*2);
  u16*   kh    = (u16*)  wsalloc((size_t)SLEN*NKVH*HDIM*2);
  u16*   vh    = (u16*)  wsalloc((size_t)SLEN*NKVH*HDIM*2);
  u16*   yh    = (u16*)  wsalloc((size_t)SLEN*HID*2);

  // 1) convert inputs/weights to bf16
  k_f32_to_bf16<<<1024, 256, 0, stream>>>(x,    xh,    SLEN*HID);
  k_f32_to_bf16<<<1024, 256, 0, stream>>>(Wqkv, wqkvh, QKVN*HID);
  k_f32_to_bf16<<<1024, 256, 0, stream>>>(Wo,   woh,   HID*HID);

  // 2) QKV projection: qkv[S, 3072] = x @ Wqkv^T
  k_gemm_bf16_nt<<<dim3(SLEN/32, QKVN/64), 32, 0, stream>>>(xh, wqkvh, qkvf,
                                                            SLEN, QKVN, HID);

  // 3) RoPE on q,k + split/convert to bf16
  k_rope_split<<<dim3(SLEN, NHQ + 2*NKVH), 32, 0, stream>>>(qkvf, fc, qh, kh, vh);

  // 4) flash attention per (head, 16-row query block)
  k_flash_attn<<<dim3(NHQ, SLEN/16), 32, 0, stream>>>(qh, kh, vh, yh);

  // 5) output projection: out[S, H] = y @ Wo^T (f32 output)
  k_gemm_bf16_nt<<<dim3(SLEN/32, HID/64), 32, 0, stream>>>(yh, woh, out,
                                                           SLEN, HID, HID);
}
